// Decoder_75076028334796
// MI455X (gfx1250) — matrix operand order repaired and checked by reading, not compiled, not run on hardware
//
#include <hip/hip_runtime.h>
#include <hip/hip_bf16.h>

// Problem constants (match reference)
#define HH 1024
#define VV 32000
#define LL 3
#define BB 64
#define TT 32

typedef __attribute__((ext_vector_type(16))) __bf16 v16bf;
typedef __attribute__((ext_vector_type(8)))  __bf16 v8bf;
typedef __attribute__((ext_vector_type(8)))  float  v8f;
typedef __attribute__((ext_vector_type(4)))  int    v4i;

typedef __attribute__((address_space(1))) v4i* gptr_v4i;   // global
typedef __attribute__((address_space(3))) v4i* lptr_v4i;   // LDS

#if __has_builtin(__builtin_amdgcn_global_load_async_to_lds_b128)
#define HAVE_ASYNC_LDS 1
#else
#define HAVE_ASYNC_LDS 0
#endif

template <int N>
__device__ __forceinline__ void wait_async() {
#if __has_builtin(__builtin_amdgcn_s_wait_asynccnt)
  __builtin_amdgcn_s_wait_asynccnt(N);
#else
  asm volatile("s_wait_asynccnt %0" ::"i"(N) : "memory");
#endif
}

#if HAVE_ASYNC_LDS
__device__ __forceinline__ void async_copy_b128(const void* gsrc, void* ldst) {
  __builtin_amdgcn_global_load_async_to_lds_b128(
      (gptr_v4i)(void*)gsrc, (lptr_v4i)(void*)ldst, 0, 0);
}
#endif

// ---------------------------------------------------------------------------
// WMMA helpers (CDNA5 16x16x32 bf16, f32 accumulate)
// A (16x32 bf16): lane = m(0..15) + 16*g ; element j -> k = g*16 + j
// B (32x16 bf16): lane holds column n = lane&15 ; element j -> k = g*16 + j
// C/D (16x16 f32): element i -> row m = i + 8*g, col n = lane&15
// ---------------------------------------------------------------------------
__device__ __forceinline__ v8f wmma_bf16(v16bf a, v16bf b, v8f c) {
  return __builtin_amdgcn_wmma_f32_16x16x32_bf16(false, a, false, b, (short)0, c, false, false);
}

__device__ __forceinline__ v16bf load_frag_a(const __bf16* __restrict__ A, int m_base,
                                             int ld, int k0, int lane) {
  const int g = lane >> 4;
  const int m = m_base + (lane & 15);
  const __bf16* p = A + (size_t)m * ld + k0 + g * 16;
  v8bf lo = *(const v8bf*)p;
  v8bf hi = *(const v8bf*)(p + 8);
  return __builtin_shufflevector(lo, hi, 0,1,2,3,4,5,6,7,8,9,10,11,12,13,14,15);
}

// B fragment straight from bf16 weight matrix W[N,K] row-major (32B contiguous)
__device__ __forceinline__ v16bf load_frag_b_bf16(const __bf16* __restrict__ W, size_t n,
                                                  int ld, int k0, int g) {
  const __bf16* p = W + n * ld + k0 + g * 16;
  v8bf lo = *(const v8bf*)p;
  v8bf hi = *(const v8bf*)(p + 8);
  return __builtin_shufflevector(lo, hi, 0,1,2,3,4,5,6,7,8,9,10,11,12,13,14,15);
}

// B fragment from f32 weight matrix (on-the-fly convert; fallback path)
__device__ __forceinline__ v16bf load_frag_b_f32(const float* __restrict__ W, size_t n,
                                                 int ld, int k0, int g) {
  const float* p = W + n * ld + k0 + g * 16;
  v8f lo = *(const v8f*)p;
  v8f hi = *(const v8f*)(p + 8);
  v16bf r;
#pragma unroll
  for (int i = 0; i < 8; ++i) { r[i] = (__bf16)lo[i]; r[i + 8] = (__bf16)hi[i]; }
  return r;
}

// B fragment from LDS-staged bf16 tile, row pitch LDB halves
__device__ __forceinline__ v16bf load_frag_b_lds(const __bf16* lds, int n, int ldb, int g) {
  const __bf16* p = lds + n * ldb + g * 16;
  v8bf lo = *(const v8bf*)p;
  v8bf hi = *(const v8bf*)(p + 8);
  return __builtin_shufflevector(lo, hi, 0,1,2,3,4,5,6,7,8,9,10,11,12,13,14,15);
}

// ---------------------------------------------------------------------------
// f32 -> bf16 weight conversion, 8 elements per thread
// ---------------------------------------------------------------------------
__global__ __launch_bounds__(256) void cvt_f32_bf16_kernel(const float* __restrict__ src,
                                                           __bf16* __restrict__ dst) {
  size_t i = ((size_t)blockIdx.x * 256 + threadIdx.x) * 8;
  v8f v = *(const v8f*)(src + i);
  v8bf o;
#pragma unroll
  for (int j = 0; j < 8; ++j) o[j] = (__bf16)v[j];
  *(v8bf*)(dst + i) = o;
}

// ---------------------------------------------------------------------------
// Embedding gather + ReLU, teacher-forced shift (ids[b,0]=0, ids[b,t]=tgt[b,t-1])
// ---------------------------------------------------------------------------
__global__ __launch_bounds__(256) void embed_relu_kernel(const int* __restrict__ tgt,
                                                         const float* __restrict__ emb,
                                                         __bf16* __restrict__ xbf) {
  int idx = blockIdx.x * 256 + threadIdx.x;      // [0, B*T*H)
  int hh  = idx & (HH - 1);
  int bt  = idx >> 10;
  int t   = bt & (TT - 1);
  int b   = bt >> 5;
  int id  = (t == 0) ? 0 : tgt[b * TT + t - 1];
  float v = emb[(size_t)id * HH + hh];
  xbf[idx] = (__bf16)fmaxf(v, 0.0f);
}

__global__ __launch_bounds__(256) void hinit_kernel(const float* __restrict__ src,
                                                    float* __restrict__ hf,
                                                    __bf16* __restrict__ hb) {
  int i = blockIdx.x * 256 + threadIdx.x;        // [0, B*H)
  float v = src[i];
  hf[i] = v;
  hb[i] = (__bf16)v;
}

#define LDB 40  // padded LDS row pitch (halves): start bank = n*20+g*8 mod 64, conflict-free

// ---------------------------------------------------------------------------
// GEMM, bf16 weights: C[2048,Ntot] = A_bf16 @ Wbf[Ntot,1024]^T (+bias)
// Weight tile (64x32 bf16 = 4KB) staged via GLOBAL_LOAD_ASYNC_TO_LDS_B128
// (each of 256 lanes DMAs its 16B segment), double-buffered in LDS so the
// next tile's DMA overlaps this tile's WMMAs. grid = (Ntot/64, 2048/256).
// ---------------------------------------------------------------------------
__global__ __launch_bounds__(256) void gemm_bf16w_kernel(const __bf16* __restrict__ A,
                                                         const __bf16* __restrict__ W,
                                                         const float* __restrict__ bias,
                                                         float* __restrict__ C, int Ntot) {
#if HAVE_ASYNC_LDS
  __shared__ __bf16 ldsB[2][64 * LDB];
#else
  __shared__ __bf16 ldsB[1][64 * LDB];
#endif
  const int tid  = threadIdx.x;
  const int lane = tid & 31, w = tid >> 5;
  const int g    = lane >> 4, nl = lane & 15;
  const int nb   = blockIdx.x * 64;
  const int mb   = blockIdx.y * 256 + w * 32;
  const int trow = tid >> 2, tk = (tid & 3) * 8;
  const __bf16* wp = W + (size_t)(nb + trow) * HH + tk;   // this lane's 16B segment

  v8f acc[2][4] = {};

#if HAVE_ASYNC_LDS
  async_copy_b128(wp, &ldsB[0][trow * LDB + tk]);
#endif

  for (int kt = 0; kt < HH / 32; ++kt) {
    const int k0 = kt * 32;
#if HAVE_ASYNC_LDS
    __syncthreads();                     // prior reads of buffer being refilled are done
    if (kt + 1 < HH / 32) {
      async_copy_b128(wp + (kt + 1) * 32, &ldsB[(kt + 1) & 1][trow * LDB + tk]);
      wait_async<1>();                   // own async for current tile retired
    } else {
      wait_async<0>();
    }
    __syncthreads();                     // all waves' DMAs for current tile visible
    const __bf16* buf = &ldsB[kt & 1][0];
#else
    v8bf w8 = *(const v8bf*)(wp + k0);
    __syncthreads();
    *(v8bf*)(&ldsB[0][trow * LDB + tk]) = w8;
    __syncthreads();
    const __bf16* buf = &ldsB[0][0];
#endif
    v16bf a0 = load_frag_a(A, mb,      HH, k0, lane);
    v16bf a1 = load_frag_a(A, mb + 16, HH, k0, lane);
#pragma unroll
    for (int nt = 0; nt < 4; ++nt) {
      v16bf b = load_frag_b_lds(buf, nt * 16 + nl, LDB, g);
      acc[0][nt] = wmma_bf16(a0, b, acc[0][nt]);
      acc[1][nt] = wmma_bf16(a1, b, acc[1][nt]);
    }
  }

#pragma unroll
  for (int mt = 0; mt < 2; ++mt)
#pragma unroll
    for (int nt = 0; nt < 4; ++nt) {
      int n = nb + nt * 16 + nl;
      float bv = bias ? bias[n] : 0.0f;
#pragma unroll
      for (int i = 0; i < 8; ++i)
        C[(size_t)(mb + mt * 16 + i + 8 * g) * Ntot + n] = acc[mt][nt][i] + bv;
    }
}

// ---------------------------------------------------------------------------
// GEMM, f32 weights (fallback when ws can't hold bf16 weights): sync LDS
// staging with f32->bf16 convert. Same tiling.
// ---------------------------------------------------------------------------
__global__ __launch_bounds__(256) void gemm_f32w_kernel(const __bf16* __restrict__ A,
                                                        const float* __restrict__ W,
                                                        const float* __restrict__ bias,
                                                        float* __restrict__ C, int Ntot) {
  __shared__ __bf16 ldsB[64 * LDB];
  const int tid  = threadIdx.x;
  const int lane = tid & 31, w = tid >> 5;
  const int g    = lane >> 4, nl = lane & 15;
  const int nb   = blockIdx.x * 64;
  const int mb   = blockIdx.y * 256 + w * 32;
  const int trow = tid >> 2, tk = (tid & 3) * 8;

  v8f acc[2][4] = {};
  const float* wp = W + (size_t)(nb + trow) * HH + tk;

  for (int k0 = 0; k0 < HH; k0 += 32) {
    v8f w8 = *(const v8f*)(wp + k0);
    __builtin_prefetch(wp + k0 + 32, 0, 0);
    __syncthreads();
    v8bf w8b;
#pragma unroll
    for (int i = 0; i < 8; ++i) w8b[i] = (__bf16)w8[i];
    *(v8bf*)(&ldsB[trow * LDB + tk]) = w8b;
    __syncthreads();

    v16bf a0 = load_frag_a(A, mb,      HH, k0, lane);
    v16bf a1 = load_frag_a(A, mb + 16, HH, k0, lane);
#pragma unroll
    for (int nt = 0; nt < 4; ++nt) {
      v16bf b = load_frag_b_lds(ldsB, nt * 16 + nl, LDB, g);
      acc[0][nt] = wmma_bf16(a0, b, acc[0][nt]);
      acc[1][nt] = wmma_bf16(a1, b, acc[1][nt]);
    }
  }

#pragma unroll
  for (int mt = 0; mt < 2; ++mt)
#pragma unroll
    for (int nt = 0; nt < 4; ++nt) {
      int n = nb + nt * 16 + nl;
      float bv = bias ? bias[n] : 0.0f;
#pragma unroll
      for (int i = 0; i < 8; ++i)
        C[(size_t)(mb + mt * 16 + i + 8 * g) * Ntot + n] = acc[mt][nt][i] + bv;
    }
}

// ---------------------------------------------------------------------------
// One GRU time step (fused hp-GEMM + gates). Each WG owns a 32-col slice of H
// and computes the r/z/n columns (j, H+j, 2H+j) so the cell update is local.
// WG = 8 waves: 4 row-tiles (B=64) x 2 col-tiles. grid = H/32 = 32.
// ---------------------------------------------------------------------------
__device__ __forceinline__ void gru_epilogue(const v8f& ar, const v8f& az, const v8f& an,
                                             const float* __restrict__ xp,
                                             const float* __restrict__ bhh,
                                             const float* __restrict__ hsrc_f,
                                             __bf16* __restrict__ hdst_bf,
                                             float* __restrict__ hdst_f,
                                             __bf16* __restrict__ ys_bf,
                                             float* __restrict__ finals,
                                             int rt, int j, int g, int t) {
  const float br_ = bhh[j], bz_ = bhh[HH + j], bn_ = bhh[2 * HH + j];
#pragma unroll
  for (int i = 0; i < 8; ++i) {
    int b = rt + i + 8 * g;
    size_t xrow = (size_t)(b * TT + t) * (3 * HH);
    float xr = xp[xrow + j], xz = xp[xrow + HH + j], xn = xp[xrow + 2 * HH + j];
    float hp = hsrc_f[(size_t)b * HH + j];
    float r  = 1.0f / (1.0f + __expf(-(xr + ar[i] + br_)));
    float z  = 1.0f / (1.0f + __expf(-(xz + az[i] + bz_)));
    float n  = tanhf(xn + r * (an[i] + bn_));
    float hn = (1.0f - z) * n + z * hp;
    hdst_f[(size_t)b * HH + j]             = hn;
    hdst_bf[(size_t)b * HH + j]            = (__bf16)hn;
    ys_bf[(size_t)(b * TT + t) * HH + j]   = (__bf16)hn;
    if (finals) finals[(size_t)b * HH + j] = hn;
  }
}

__global__ __launch_bounds__(256) void gru_step_bf16w_kernel(
    const __bf16* __restrict__ hsrc_bf, const float* __restrict__ hsrc_f,
    __bf16* __restrict__ hdst_bf, float* __restrict__ hdst_f,
    const float* __restrict__ xp, const __bf16* __restrict__ Whh,
    const float* __restrict__ bhh, __bf16* __restrict__ ys_bf,
    float* __restrict__ finals, int t) {
  const int tid  = threadIdx.x;
  const int lane = tid & 31, w = tid >> 5;
  const int g    = lane >> 4, nl = lane & 15;
  const int rt   = (w & 3) * 16;
  const int j0   = blockIdx.x * 32 + (w >> 2) * 16;

  v8f ar = {}, az = {}, an = {};
  for (int k0 = 0; k0 < HH; k0 += 32) {
    v16bf a  = load_frag_a(hsrc_bf, rt, HH, k0, lane);
    v16bf br = load_frag_b_bf16(Whh, (size_t)(j0 + nl),          HH, k0, g);
    v16bf bz = load_frag_b_bf16(Whh, (size_t)(HH + j0 + nl),     HH, k0, g);
    v16bf bn = load_frag_b_bf16(Whh, (size_t)(2 * HH + j0 + nl), HH, k0, g);
    ar = wmma_bf16(a, br, ar);
    az = wmma_bf16(a, bz, az);
    an = wmma_bf16(a, bn, an);
  }
  gru_epilogue(ar, az, an, xp, bhh, hsrc_f, hdst_bf, hdst_f, ys_bf, finals,
               rt, j0 + nl, g, t);
}

__global__ __launch_bounds__(256) void gru_step_f32w_kernel(
    const __bf16* __restrict__ hsrc_bf, const float* __restrict__ hsrc_f,
    __bf16* __restrict__ hdst_bf, float* __restrict__ hdst_f,
    const float* __restrict__ xp, const float* __restrict__ Whh,
    const float* __restrict__ bhh, __bf16* __restrict__ ys_bf,
    float* __restrict__ finals, int t) {
  const int tid  = threadIdx.x;
  const int lane = tid & 31, w = tid >> 5;
  const int g    = lane >> 4, nl = lane & 15;
  const int rt   = (w & 3) * 16;
  const int j0   = blockIdx.x * 32 + (w >> 2) * 16;

  v8f ar = {}, az = {}, an = {};
  for (int k0 = 0; k0 < HH; k0 += 32) {
    v16bf a  = load_frag_a(hsrc_bf, rt, HH, k0, lane);
    v16bf br = load_frag_b_f32(Whh, (size_t)(j0 + nl),          HH, k0, g);
    v16bf bz = load_frag_b_f32(Whh, (size_t)(HH + j0 + nl),     HH, k0, g);
    v16bf bn = load_frag_b_f32(Whh, (size_t)(2 * HH + j0 + nl), HH, k0, g);
    ar = wmma_bf16(a, br, ar);
    az = wmma_bf16(a, bz, az);
    an = wmma_bf16(a, bn, an);
  }
  gru_epilogue(ar, az, an, xp, bhh, hsrc_f, hdst_bf, hdst_f, ys_bf, finals,
               rt, j0 + nl, g, t);
}

// ---------------------------------------------------------------------------
// In-place log_softmax over V=32000 per row (online max/sum + WG reduction)
// ---------------------------------------------------------------------------
__global__ __launch_bounds__(256) void logsoftmax_kernel(float* __restrict__ out) {
  __shared__ float sm[256], ss[256];
  const int tid = threadIdx.x;
  float* p = out + (size_t)blockIdx.x * VV;

  float m = -1e30f, s = 0.0f;
  for (int i = tid; i < VV; i += 256) {
    float x = p[i];
    if (x > m) { s = s * __expf(m - x) + 1.0f; m = x; }
    else       { s += __expf(x - m); }
  }
  sm[tid] = m; ss[tid] = s;
  __syncthreads();
  for (int off = 128; off > 0; off >>= 1) {
    if (tid < off) {
      float m2 = sm[tid + off], s2 = ss[tid + off];
      float mm = fmaxf(sm[tid], m2);
      ss[tid] = ss[tid] * __expf(sm[tid] - mm) + s2 * __expf(m2 - mm);
      sm[tid] = mm;
    }
    __syncthreads();
  }
  const float gl = sm[0] + __logf(ss[0]);
  for (int i = tid; i < VV; i += 256) p[i] = p[i] - gl;
}

// ---------------------------------------------------------------------------
extern "C" void kernel_launch(void* const* d_in, const int* in_sizes, int n_in,
                              void* d_out, int out_size, void* d_ws, size_t ws_size,
                              hipStream_t stream) {
  (void)in_sizes; (void)n_in; (void)out_size;
  const float* enc_hid = (const float*)d_in[1];   // [L,B,H] (d_in[0] unused)
  const int*   tgt     = (const int*)  d_in[2];   // [B,T]
  const float* emb     = (const float*)d_in[3];   // [V,H]
  const float* W_ih    = (const float*)d_in[4];   // [L,3H,H]
  const float* W_hh    = (const float*)d_in[5];   // [L,3H,H]
  const float* b_ih    = (const float*)d_in[6];   // [L,3H]
  const float* b_hh    = (const float*)d_in[7];   // [L,3H]
  const float* W_out   = (const float*)d_in[8];   // [V,H]
  const float* b_out   = (const float*)d_in[9];   // [V]

  float* logits = (float*)d_out;                         // [B*T, V]
  float* finals = logits + (size_t)BB * TT * VV;         // [L, B, H]

  // workspace carve-up
  char* ws = (char*)d_ws;
  size_t off = 0;
  auto alloc = [&](size_t bytes) { void* p = ws + off; off = (off + bytes + 255) & ~(size_t)255; return p; };
  __bf16* x_bf  = (__bf16*)alloc((size_t)BB * TT * HH * 2);
  float*  xp    = (float*) alloc((size_t)BB * TT * 3 * HH * 4);
  float*  hf[2]; __bf16* hb[2];
  hf[0] = (float*) alloc((size_t)BB * HH * 4);
  hf[1] = (float*) alloc((size_t)BB * HH * 4);
  hb[0] = (__bf16*)alloc((size_t)BB * HH * 2);
  hb[1] = (__bf16*)alloc((size_t)BB * HH * 2);
  // optional bf16 weight mirrors (preferred path)
  const size_t nWih = (size_t)LL * 3 * HH * HH;   // 9,437,184
  const size_t nWhh = nWih;
  const size_t nWo  = (size_t)VV * HH;            // 32,768,000
  __bf16* wih_bf  = (__bf16*)alloc(nWih * 2);
  __bf16* whh_bf  = (__bf16*)alloc(nWhh * 2);
  __bf16* wout_bf = (__bf16*)alloc(nWo * 2);
  const bool bf16w = (ws_size >= off);            // constant per harness => deterministic

  if (bf16w) {
    cvt_f32_bf16_kernel<<<(int)(nWih / 2048), 256, 0, stream>>>(W_ih, wih_bf);
    cvt_f32_bf16_kernel<<<(int)(nWhh / 2048), 256, 0, stream>>>(W_hh, whh_bf);
    cvt_f32_bf16_kernel<<<(int)(nWo  / 2048), 256, 0, stream>>>(W_out, wout_bf);
  }

  // 1) embedding + relu -> x_bf16
  embed_relu_kernel<<<(BB * TT * HH) / 256, 256, 0, stream>>>(tgt, emb, x_bf);

  for (int l = 0; l < LL; ++l) {
    // 2) xp = x @ W_ih[l]^T + b_ih[l]   (2048 x 3072, K=1024)
    dim3 gXP((3 * HH) / 64, (BB * TT) / 256);
    if (bf16w)
      gemm_bf16w_kernel<<<gXP, 256, 0, stream>>>(
          x_bf, wih_bf + (size_t)l * 3 * HH * HH, b_ih + (size_t)l * 3 * HH, xp, 3 * HH);
    else
      gemm_f32w_kernel<<<gXP, 256, 0, stream>>>(
          x_bf, W_ih + (size_t)l * 3 * HH * HH, b_ih + (size_t)l * 3 * HH, xp, 3 * HH);

    // 3) h0 = encoder_hidden[l]
    hinit_kernel<<<(BB * HH) / 256, 256, 0, stream>>>(
        enc_hid + (size_t)l * BB * HH, hf[0], hb[0]);

    // 4) sequential GRU steps (ping-pong h; ys overwrites x_bf in place)
    for (int t = 0; t < TT; ++t) {
      int src = t & 1, dst = (t + 1) & 1;
      float* fin = (t == TT - 1) ? (finals + (size_t)l * BB * HH) : nullptr;
      if (bf16w)
        gru_step_bf16w_kernel<<<HH / 32, 256, 0, stream>>>(
            hb[src], hf[src], hb[dst], hf[dst], xp,
            whh_bf + (size_t)l * 3 * HH * HH, b_hh + (size_t)l * 3 * HH, x_bf, fin, t);
      else
        gru_step_f32w_kernel<<<HH / 32, 256, 0, stream>>>(
            hb[src], hf[src], hb[dst], hf[dst], xp,
            W_hh + (size_t)l * 3 * HH * HH, b_hh + (size_t)l * 3 * HH, x_bf, fin, t);
    }
  }

  // 5) logits = x @ W_out^T + b_out   (2048 x 32000, K=1024)
  dim3 gOUT(VV / 64, (BB * TT) / 256);
  if (bf16w)
    gemm_bf16w_kernel<<<gOUT, 256, 0, stream>>>(x_bf, wout_bf, b_out, logits, VV);
  else
    gemm_f32w_kernel<<<gOUT, 256, 0, stream>>>(x_bf, W_out, b_out, logits, VV);

  // 6) in-place log_softmax
  logsoftmax_kernel<<<BB * TT, 256, 0, stream>>>(logits);
}
